// MixtureOfTokensLayer_34093450395857
// MI455X (gfx1250) — matrix operand order repaired and checked by reading, not compiled, not run on hardware
//
#include <hip/hip_runtime.h>
#include <hip/hip_bf16.h>
#include <stdint.h>

// ---------------------------------------------------------------------------
// Mixture-of-Tokens layer for MI455X (gfx1250), wave32 + WMMA bf16.
//   Phase 0: convert+transpose weights fp32 -> bf16 (lin1T[e][f][d], lin2T[e][d][f])
//   Phase 1: router logits + softmax over the 32-token group + merge -> merged bf16
//   Phase 2a: h = relu(merged @ lin1)   (8x GEMM 2048x512x1024, v_wmma_f32_16x16x32_bf16)
//   Phase 2b: eo = h @ lin2             (8x GEMM 2048x1024x512, fp32 out)
//   Phase 3: out = emit_w @ eo          (memory-bound weighted scatter back to tokens)
// GEMM: 64x64 tile per wave (16 WMMA / 16 b128 loads per K-step, 32 flops/B from
// the 192MB L2-resident working set), block tile 128x256 with 8 waves.
// ---------------------------------------------------------------------------

typedef __attribute__((ext_vector_type(16))) __bf16 v16bf;
typedef __attribute__((ext_vector_type(8)))  float  v8f;

constexpr int D_MODEL = 1024;
constexpr int EXPERT_SIZE = 512;
constexpr int N_EXPERTS = 8;
constexpr int GROUP = 32;   // == BATCH
constexpr int SEQ = 2048;

#define DEVFN static __device__ __forceinline__

union FragBF { v16bf v; uint4 u[2]; };

// ---------------------------------------------------------------------------
// Phase 0: dst[e][c][r] = (bf16) src[e][r][c]
// ---------------------------------------------------------------------------
__global__ __launch_bounds__(256) void transpose_f32_to_bf16(
    const float* __restrict__ src, __bf16* __restrict__ dst, int R, int Ccols)
{
  const size_t per = (size_t)R * Ccols;
  const size_t e = blockIdx.y;
  src += e * per;
  dst += e * per;
  size_t idx = (size_t)blockIdx.x * 256 + threadIdx.x;  // idx = c*R + r
  if (idx < per) {
    int c = (int)(idx / R);
    int r = (int)(idx % R);
    dst[idx] = (__bf16)src[(size_t)r * Ccols + c];
  }
}

// ---------------------------------------------------------------------------
// Phase 1: one block per sequence position s.
//   logits[k][e] = dot(x[k,s,:], controller[:,e])   (wave-parallel over d)
//   w[k][e]      = softmax over k (the 32-token group) per expert
//   merged[s][e][:] = sum_k w[k][e] * x[k,s,:]      -> bf16 for WMMA phase
// ---------------------------------------------------------------------------
__global__ __launch_bounds__(256) void router_merge(
    const float* __restrict__ x, const float* __restrict__ ctrl,
    __bf16* __restrict__ merged, float* __restrict__ mw_out)
{
  const int s = blockIdx.x;
  const int tid = threadIdx.x;
  const int lane = tid & 31;
  const int w = tid >> 5;           // wave id, 0..7

  __shared__ float lg[GROUP * N_EXPERTS];    // logits 32x8
  __shared__ float wsm[GROUP * N_EXPERTS];   // softmax weights 32x8

  // --- logits: wave w handles tokens 4w .. 4w+3, lanes split the d dimension ---
  for (int t = 0; t < 4; ++t) {
    const int k = w * 4 + t;
    const float* xrow = x + (size_t)k * SEQ * D_MODEL + (size_t)s * D_MODEL;
    float acc[8] = {0.f, 0.f, 0.f, 0.f, 0.f, 0.f, 0.f, 0.f};
    for (int d = lane; d < D_MODEL; d += 32) {
      float xv = xrow[d];
      const float4* cp = (const float4*)(ctrl + (size_t)d * N_EXPERTS);
      float4 c0 = cp[0], c1 = cp[1];
      acc[0] += xv * c0.x; acc[1] += xv * c0.y;
      acc[2] += xv * c0.z; acc[3] += xv * c0.w;
      acc[4] += xv * c1.x; acc[5] += xv * c1.y;
      acc[6] += xv * c1.z; acc[7] += xv * c1.w;
    }
#pragma unroll
    for (int e = 0; e < N_EXPERTS; ++e) {
      float v = acc[e];
      for (int o = 16; o > 0; o >>= 1) v += __shfl_xor(v, o, 32);
      if (lane == 0) lg[k * N_EXPERTS + e] = v;
    }
  }
  __syncthreads();

  // --- softmax over k per expert: wave 0, lane = token ---
  if (w == 0) {
    const int k = lane;
#pragma unroll
    for (int e = 0; e < N_EXPERTS; ++e) {
      float v = lg[k * N_EXPERTS + e];
      float m = v;
      for (int o = 16; o > 0; o >>= 1) m = fmaxf(m, __shfl_xor(m, o, 32));
      float ex = __expf(v - m);
      float sm = ex;
      for (int o = 16; o > 0; o >>= 1) sm += __shfl_xor(sm, o, 32);
      float ww = ex / sm;
      wsm[k * N_EXPERTS + e] = ww;
      mw_out[(size_t)s * (GROUP * N_EXPERTS) + k * N_EXPERTS + e] = ww;
    }
  }
  __syncthreads();

  // --- merged[s][e][d] = sum_k wsm[k][e] * x[k][s][d] ---
#pragma unroll
  for (int j = 0; j < 4; ++j) {
    const int d = tid + j * 256;
    float acc[8] = {0.f, 0.f, 0.f, 0.f, 0.f, 0.f, 0.f, 0.f};
    for (int k = 0; k < GROUP; ++k) {
      float xv = x[(size_t)k * SEQ * D_MODEL + (size_t)s * D_MODEL + d];
#pragma unroll
      for (int e = 0; e < N_EXPERTS; ++e) acc[e] += wsm[k * N_EXPERTS + e] * xv;
    }
#pragma unroll
    for (int e = 0; e < N_EXPERTS; ++e)
      merged[((size_t)s * N_EXPERTS + e) * D_MODEL + d] = (__bf16)acc[e];
  }
}

// ---------------------------------------------------------------------------
// Phase 2: batched-per-expert GEMM with v_wmma_f32_16x16x32_bf16.
// Block = 8 waves (2x4), each wave owns a 64x64 C tile -> block tile 128x256.
// A: row-major (m,k). Bt: B^T row-major (n,k). Working set is L2-resident.
// A-frag per ISA: lanes 0-15 row M=lane, K={kb..kb+7, kb+16..kb+23}, kb=k0 (+8 hi lanes)
// B-frag:         lanes 0-15 col N=lane, K={kb..kb+7, kb+16..kb+23}, kb=k0 (+8 hi lanes)
// ---------------------------------------------------------------------------
template <bool RELU, typename OutT>
__global__ __launch_bounds__(256) void expert_gemm(
    const __bf16* __restrict__ Abase, const __bf16* __restrict__ Btbase,
    OutT* __restrict__ Cbase, int K,
    int lda, int ldb, int ldc,
    size_t eStrideA, size_t eStrideB, size_t eStrideC)
{
  const int e = blockIdx.z;
  const __bf16* A  = Abase  + eStrideA * (size_t)e;
  const __bf16* Bt = Btbase + eStrideB * (size_t)e;
  OutT* C = Cbase + eStrideC * (size_t)e;

  const int tid  = threadIdx.x;
  const int lane = tid & 31;
  const int wave = tid >> 5;
  const int wm = wave >> 2;   // 0..1
  const int wn = wave & 3;    // 0..3

  const int m0 = blockIdx.x * 128 + wm * 64;
  const int n0 = blockIdx.y * 256 + wn * 64;

  // Per-lane fragment base pointers, hoisted out of the K loop.
  const int lr  = lane & 15;
  const int kha = (lane >> 4) << 3;   // +8  elements for upper half-wave (A)
  const int khb = (lane >> 4) << 3;   // +8  elements for upper half-wave (B)
  const __bf16* pa[4];
  const __bf16* pb[4];
#pragma unroll
  for (int i = 0; i < 4; ++i) {
    pa[i] = A  + (size_t)(m0 + 16 * i + lr) * lda + kha;
    pb[i] = Bt + (size_t)(n0 + 16 * i + lr) * ldb + khb;
  }

  v8f c[4][4] = {};

  for (int k0 = 0; k0 < K; k0 += 32) {
    FragBF fa[4], fb[4];
#pragma unroll
    for (int i = 0; i < 4; ++i) {
      fa[i].u[0] = *(const uint4*)(pa[i] + k0);
      fa[i].u[1] = *(const uint4*)(pa[i] + k0 + 16);
      fb[i].u[0] = *(const uint4*)(pb[i] + k0);
      fb[i].u[1] = *(const uint4*)(pb[i] + k0 + 16);
      // global_prefetch_b8: pull the stream 8 K-steps (512B) ahead into cache.
      __builtin_prefetch(pa[i] + k0 + 256, 0, 3);
      __builtin_prefetch(pb[i] + k0 + 256, 0, 3);
    }
#pragma unroll
    for (int mi = 0; mi < 4; ++mi)
#pragma unroll
      for (int ni = 0; ni < 4; ++ni)
        c[mi][ni] = __builtin_amdgcn_wmma_f32_16x16x32_bf16(
            false, fa[mi].v, false, fb[ni].v, (short)0, c[mi][ni], false, false);
  }

  // C/D layout: VGPR r: lanes 0-15 -> (M=r, N=lane); lanes 16-31 -> (M=8+r, N=lane-16)
  const int nc = lane & 15;
  const int mb = (lane >> 4) << 3;
#pragma unroll
  for (int mi = 0; mi < 4; ++mi) {
#pragma unroll
    for (int ni = 0; ni < 4; ++ni) {
#pragma unroll
      for (int r = 0; r < 8; ++r) {
        float v = c[mi][ni][r];
        if (RELU) v = v > 0.f ? v : 0.f;
        C[(size_t)(m0 + mi * 16 + mb + r) * ldc + (n0 + ni * 16 + nc)] = (OutT)v;
      }
    }
  }
}

// ---------------------------------------------------------------------------
// Phase 3: out[b][s][d] = sum_e w[s][b][e] * eo[s][e][d]
// ---------------------------------------------------------------------------
__global__ __launch_bounds__(256) void emit_tokens(
    const float* __restrict__ eo, const float* __restrict__ mw,
    float* __restrict__ out)
{
  const int s = blockIdx.x;
  const int tid = threadIdx.x;

  __shared__ float eos[N_EXPERTS * D_MODEL];   // 32 KB
  __shared__ float wv[GROUP * N_EXPERTS];      // 1 KB

  for (int i = tid; i < N_EXPERTS * D_MODEL; i += 256)
    eos[i] = eo[(size_t)s * (N_EXPERTS * D_MODEL) + i];
  wv[tid] = mw[(size_t)s * (GROUP * N_EXPERTS) + tid];
  __syncthreads();

  for (int b = 0; b < GROUP; ++b) {
    float wb[8];
#pragma unroll
    for (int e = 0; e < N_EXPERTS; ++e) wb[e] = wv[b * N_EXPERTS + e];
#pragma unroll
    for (int j = 0; j < 4; ++j) {
      const int d = tid + j * 256;
      float acc = 0.f;
#pragma unroll
      for (int e = 0; e < N_EXPERTS; ++e) acc += wb[e] * eos[e * D_MODEL + d];
      out[((size_t)b * SEQ + s) * D_MODEL + d] = acc;
    }
  }
}

// ---------------------------------------------------------------------------
extern "C" void kernel_launch(void* const* d_in, const int* in_sizes, int n_in,
                              void* d_out, int out_size, void* d_ws, size_t ws_size,
                              hipStream_t stream)
{
  const float* x    = (const float*)d_in[0];   // (32, 2048, 1024)
  const float* lin1 = (const float*)d_in[1];   // (8, 1024, 512)
  const float* lin2 = (const float*)d_in[2];   // (8, 512, 1024)
  const float* ctrl = (const float*)d_in[3];   // (1024, 8)
  float* out = (float*)d_out;

  char* ws = (char*)d_ws;
  size_t off = 0;
  __bf16* lin1T  = (__bf16*)(ws + off); off += (size_t)N_EXPERTS * EXPERT_SIZE * D_MODEL * 2;  // [e][f][d]
  __bf16* lin2T  = (__bf16*)(ws + off); off += (size_t)N_EXPERTS * D_MODEL * EXPERT_SIZE * 2;  // [e][d][f]
  __bf16* merged = (__bf16*)(ws + off); off += (size_t)SEQ * N_EXPERTS * D_MODEL * 2;          // [s][e][d]
  __bf16* hbuf   = (__bf16*)(ws + off); off += (size_t)SEQ * N_EXPERTS * EXPERT_SIZE * 2;      // [s][e][f]
  float*  eo     = (float*) (ws + off); off += (size_t)SEQ * N_EXPERTS * D_MODEL * 4;          // [s][e][d]
  float*  mw     = (float*) (ws + off); off += (size_t)SEQ * GROUP * N_EXPERTS * 4;            // [s][k][e]
  if (ws_size < off) return;  // ~130 MB of scratch required

  // Phase 0: weight convert + transpose (524288 elements per expert -> 2048 blocks)
  dim3 tgrid(2048, N_EXPERTS);
  transpose_f32_to_bf16<<<tgrid, 256, 0, stream>>>(lin1, lin1T, D_MODEL, EXPERT_SIZE);
  transpose_f32_to_bf16<<<tgrid, 256, 0, stream>>>(lin2, lin2T, EXPERT_SIZE, D_MODEL);

  // Phase 1: routing + merge
  router_merge<<<SEQ, 256, 0, stream>>>(x, ctrl, merged, mw);

  // Phase 2a: h = relu(merged @ lin1)  per expert: M=2048, N=512, K=1024
  //   A = merged (lda = 8*1024, expert offset e*1024), Bt = lin1T (ldb = 1024),
  //   C = h bf16 (ldc = 8*512, expert offset e*512)
  expert_gemm<true, __bf16><<<dim3(SEQ / 128, EXPERT_SIZE / 256, N_EXPERTS), 256, 0, stream>>>(
      merged, lin1T, hbuf, D_MODEL,
      N_EXPERTS * D_MODEL, D_MODEL, N_EXPERTS * EXPERT_SIZE,
      (size_t)D_MODEL, (size_t)EXPERT_SIZE * D_MODEL, (size_t)EXPERT_SIZE);

  // Phase 2b: eo = h @ lin2  per expert: M=2048, N=1024, K=512
  expert_gemm<false, float><<<dim3(SEQ / 128, D_MODEL / 256, N_EXPERTS), 256, 0, stream>>>(
      hbuf, lin2T, eo, EXPERT_SIZE,
      N_EXPERTS * EXPERT_SIZE, EXPERT_SIZE, N_EXPERTS * D_MODEL,
      (size_t)EXPERT_SIZE, (size_t)D_MODEL * EXPERT_SIZE, (size_t)D_MODEL);

  // Phase 3: emit back to tokens
  emit_tokens<<<SEQ, 256, 0, stream>>>(eo, mw, out);
}
